// TokenformerAttention_4741643895455
// MI455X (gfx1250) — matrix operand order repaired and checked by reading, compiled, not run on hardware
//
#include <hip/hip_runtime.h>
#include <cmath>

typedef __attribute__((ext_vector_type(16))) _Float16 v16h;
typedef __attribute__((ext_vector_type(8)))  _Float16 v8h;
typedef __attribute__((ext_vector_type(8)))  float    v8f;

typedef __attribute__((ext_vector_type(4))) unsigned int tdm_v4u;
typedef __attribute__((ext_vector_type(8))) int          tdm_v8i;
typedef __attribute__((ext_vector_type(4))) int          tdm_v4i;

namespace {
constexpr int kDim   = 1024;
constexpr int kHeads = 16;
constexpr int kDh    = 64;
constexpr int kInner = 1024;
constexpr int kP     = 1024;
constexpr int kB     = 4;
constexpr int kN     = 2048;
constexpr int kM     = kB * kN;   // 8192 token rows
}

union FragCast { v16h v; v8h h[2]; };

// A fragment (16x32 f16, M=lane&15). Per ISA 7.12.2:
// lanes 0-15: halves 0..7 = K0..7,  halves 8..15 = K16..23
// lanes 16-31: halves 0..7 = K8..15, halves 8..15 = K24..31
__device__ __forceinline__ v16h load_frag_a(const _Float16* base, int ld) {
  const int lane = threadIdx.x & 31;
  const int row  = lane & 15;
  const int hi   = lane >> 4;
  const _Float16* p = base + (size_t)row * ld;
  FragCast f;
  f.h[0] = *(const v8h*)(p + hi * 8);
  f.h[1] = *(const v8h*)(p + 16 + hi * 8);
  return f.v;
}

// B fragment (32x16 f16) read from a K-contiguous "Bt" matrix (row n holds K).
// lanes 0-15: N=lane, K=0..7,16..23 ; lanes 16-31: N=lane-16, K=8..15,24..31
__device__ __forceinline__ v16h load_frag_b(const _Float16* base, int ld) {
  const int lane = threadIdx.x & 31;
  const int col  = lane & 15;
  const int hi   = lane >> 4;
  const _Float16* p = base + (size_t)col * ld;
  FragCast f;
  f.h[0] = *(const v8h*)(p + hi * 8);
  f.h[1] = *(const v8h*)(p + 16 + hi * 8);
  return f.v;
}

__device__ __forceinline__ v8f wmma_f16(v16h a, v16h b, v8f c) {
  return __builtin_amdgcn_wmma_f32_16x16x32_f16(false, a, false, b, (short)0, c,
                                                false, false);
}

// ds_swizzle XOR butterfly (single ds op; stays within 16-lane half for M<16)
template <int MASK>
__device__ __forceinline__ float xor_fmax(float x) {
  const int w = __builtin_amdgcn_ds_swizzle(__float_as_int(x), 0x1f | (MASK << 10));
  return fmaxf(x, __int_as_float(w));
}
template <int MASK>
__device__ __forceinline__ float xor_fadd(float x) {
  const int w = __builtin_amdgcn_ds_swizzle(__float_as_int(x), 0x1f | (MASK << 10));
  return x + __int_as_float(w);
}

__device__ __forceinline__ unsigned lds_off(const void* p) {
  // Generic LDS-aperture address: low 32 bits are the LDS byte offset.
  return (unsigned)(unsigned long long)(uintptr_t)p;
}

// ------------------------------------------------ Tensor Data Mover (TDM)
// 2D tile load: tile_d1 rows of tile_d0 elements (f16), row stride stride_d0
// elements, packed contiguously into LDS at lds_byte_off. D# per ISA §8.3-8.4.
__device__ __forceinline__ void tdm_load_2d(unsigned lds_byte_off,
                                            const void* global_ptr,
                                            unsigned tile_d0, unsigned tile_d1,
                                            unsigned stride_d0) {
  const unsigned long long ga = (unsigned long long)(uintptr_t)global_ptr;
  tdm_v4u g0;
  g0.x = 1u;                                                  // count=1 (valid D#)
  g0.y = lds_byte_off;                                        // lds_addr
  g0.z = (unsigned)ga;                                        // global_addr[31:0]
  g0.w = (unsigned)((ga >> 32) & 0x01ffffffu) | (2u << 30);   // addr[56:32] | type=2
  tdm_v8i g1;
  g1[0] = (int)(1u << 16);                                    // data_size=1 -> 2 bytes
  g1[1] = (int)((tile_d0 & 0xffffu) << 16);                   // tensor_dim0[15:0]
  g1[2] = (int)(((tile_d0 >> 16) & 0xffffu) |
                ((tile_d1 & 0xffffu) << 16));                 // dim0 hi | dim1 lo
  g1[3] = (int)(((tile_d1 >> 16) & 0xffffu) |
                ((tile_d0 & 0xffffu) << 16));                 // dim1 hi | tile_dim0
  g1[4] = (int)(tile_d1 & 0xffffu);                           // tile_dim1 (tile_dim2=0)
  g1[5] = (int)stride_d0;                                     // tensor_dim0_stride lo
  g1[6] = 0;
  g1[7] = 0;
  const tdm_v4i z4 = {0, 0, 0, 0};
#if __clang_major__ >= 23
  const tdm_v8i z8 = {0, 0, 0, 0, 0, 0, 0, 0};
  __builtin_amdgcn_tensor_load_to_lds(g0, g1, z4, z4, z8, 0);
#else
  __builtin_amdgcn_tensor_load_to_lds(g0, g1, z4, z4, 0);
#endif
}

// ---------------------------------------------------------------- LayerNorm
__global__ void ln_to_f16_kernel(const float* __restrict__ x,
                                 const float* __restrict__ gamma,
                                 const float* __restrict__ beta,
                                 _Float16* __restrict__ out) {
  __shared__ float red[256];
  const int row = blockIdx.x;
  const int t   = threadIdx.x;
  const float* xr = x + (size_t)row * kDim;
  float v[4];
  float s = 0.f;
#pragma unroll
  for (int i = 0; i < 4; ++i) { v[i] = xr[t + i * 256]; s += v[i]; }
  red[t] = s; __syncthreads();
  for (int k = 128; k > 0; k >>= 1) { if (t < k) red[t] += red[t + k]; __syncthreads(); }
  const float mean = red[0] * (1.f / kDim);
  __syncthreads();
  float s2 = 0.f;
#pragma unroll
  for (int i = 0; i < 4; ++i) { const float d = v[i] - mean; s2 += d * d; }
  red[t] = s2; __syncthreads();
  for (int k = 128; k > 0; k >>= 1) { if (t < k) red[t] += red[t + k]; __syncthreads(); }
  const float rstd = rsqrtf(red[0] * (1.f / kDim) + 1e-5f);
  _Float16* orow = out + (size_t)row * kDim;
#pragma unroll
  for (int i = 0; i < 4; ++i) {
    const int c = t + i * 256;
    orow[c] = (_Float16)((v[i] - mean) * rstd * gamma[c] + beta[c]);
  }
}

// -------------------------------------------------- row L2 normalize + GELU
__global__ void l2norm_gelu_kernel(const float* __restrict__ sim,
                                   _Float16* __restrict__ out) {
  __shared__ float red[256];
  const int row = blockIdx.x;
  const int t   = threadIdx.x;
  const float* sr = sim + (size_t)row * kP;
  float v[4];
  float s2 = 0.f;
#pragma unroll
  for (int i = 0; i < 4; ++i) { v[i] = sr[t + i * 256]; s2 += v[i] * v[i]; }
  red[t] = s2; __syncthreads();
  for (int k = 128; k > 0; k >>= 1) { if (t < k) red[t] += red[t + k]; __syncthreads(); }
  const float inv = 1.f / fmaxf(sqrtf(red[0]), 1e-12f);
  _Float16* orow = out + (size_t)row * kP;
#pragma unroll
  for (int i = 0; i < 4; ++i) {
    const float xv = v[i] * inv;
    const float y  = 0.5f * xv * (1.f + erff(xv * 0.70710678118654752f));  // exact-erf GELU
    orow[t + i * 256] = (_Float16)y;
  }
}

// ------------------------------------------------------ param preprocessing
__global__ void cvt_f32_f16_kernel(const float* __restrict__ in,
                                   _Float16* __restrict__ out, int n) {
  const int i = blockIdx.x * blockDim.x + threadIdx.x;
  if (i < n) out[i] = (_Float16)in[i];
}

// [P, E] f32 -> [E, P] f16 (makes val-matmuls K-contiguous / NT)
__global__ void transpose_cvt_kernel(const float* __restrict__ in,
                                     _Float16* __restrict__ out) {
  const int i = blockIdx.x * blockDim.x + threadIdx.x;  // 1M elements
  const int e = i >> 10, p = i & 1023;
  out[i] = (_Float16)in[(size_t)p * 1024 + e];
}

// --------------------------------------------------------------- NT GEMM
// C[M,1024] = A[M,1024] * Bt[1024,1024]^T  (f16 K-contiguous operands, f32 acc)
// TDM double-buffers 64x64 A and B tiles into LDS; 4 waves consume fragments.
// MODE 0: store f32 row-major; MODE 1: f16 row-major;
// MODE 2: f16 transposed per head into vt[bh][d][n] (for PV matmul).
template <int MODE>
__global__ void gemm_nt_kernel(const _Float16* __restrict__ A,
                               const _Float16* __restrict__ Bt,
                               void* __restrict__ C) {
  constexpr int K = 1024, Nn = 1024;
  __shared__ __attribute__((aligned(16))) _Float16 abuf[2][64 * 64];
  __shared__ __attribute__((aligned(16))) _Float16 bbuf[2][64 * 64];
  const int wave = threadIdx.x >> 5;
  const int lane = threadIdx.x & 31;
  const int row0 = blockIdx.y * 64;
  const int col0 = blockIdx.x * 64;
  const _Float16* aTile = A + (size_t)row0 * K;
  const _Float16* bTile = Bt + (size_t)col0 * K;
  v8f acc[4] = {};
  constexpr int NKT = K / 64;   // 16 K-steps
  if (wave == 0) {
    tdm_load_2d(lds_off(&abuf[0][0]), aTile, 64, 64, K);
    tdm_load_2d(lds_off(&bbuf[0][0]), bTile, 64, 64, K);
  }
  for (int t = 0; t < NKT; ++t) {
    const int cur = t & 1;
    if (wave == 0) {
      if (t + 1 < NKT) {
        tdm_load_2d(lds_off(&abuf[cur ^ 1][0]), aTile + (t + 1) * 64, 64, 64, K);
        tdm_load_2d(lds_off(&bbuf[cur ^ 1][0]), bTile + (t + 1) * 64, 64, 64, K);
        __builtin_amdgcn_s_wait_tensorcnt(2);   // pair t complete
      } else {
        __builtin_amdgcn_s_wait_tensorcnt(0);
      }
    }
    __syncthreads();   // publish LDS tile t
#pragma unroll
    for (int ks = 0; ks < 64; ks += 32) {
      const v16h a = load_frag_a(&abuf[cur][wave * 16 * 64 + ks], 64);
#pragma unroll
      for (int nt = 0; nt < 4; ++nt) {
        const v16h bf = load_frag_b(&bbuf[cur][nt * 16 * 64 + ks], 64);
        acc[nt] = wmma_f16(a, bf, acc[nt]);
      }
    }
    __syncthreads();   // all waves done with tile t before TDM overwrites it
  }
  const int hi = lane >> 4, cl = lane & 15;
#pragma unroll
  for (int nt = 0; nt < 4; ++nt) {
#pragma unroll
    for (int i = 0; i < 8; ++i) {
      const int r = row0 + wave * 16 + i + 8 * hi;  // C layout: VGPR i -> row
      const int c = col0 + nt * 16 + cl;            // N = lane&15
      const float val = acc[nt][i];
      if (MODE == 0) {
        ((float*)C)[(size_t)r * Nn + c] = val;
      } else if (MODE == 1) {
        ((_Float16*)C)[(size_t)r * Nn + c] = (_Float16)val;
      } else {
        const int bt  = r >> 11;                    // batch
        const int tok = r & (kN - 1);
        const int h   = c >> 6;
        const int d   = c & 63;
        ((_Float16*)C)[((size_t)(bt * kHeads + h) * kDh + d) * kN + tok] = (_Float16)val;
      }
    }
  }
}

// ------------------------------------------------- flash softmax attention
// 1 wave = 16 query rows of one (b,h); 4 waves share TDM-staged K/V tiles.
// 32-key tiles, online max/sum; P goes through LDS to become an A fragment.
__global__ void flash_attn_kernel(const _Float16* __restrict__ q,
                                  const _Float16* __restrict__ k,
                                  const _Float16* __restrict__ vt,
                                  _Float16* __restrict__ out) {
  __shared__ __attribute__((aligned(16))) _Float16 kbuf[2][32 * 64];  // [j][d]
  __shared__ __attribute__((aligned(16))) _Float16 vbuf[2][64 * 32];  // [d][j]
  __shared__ __attribute__((aligned(16))) _Float16 pbuf[4][16 * 32];
  const int wave = threadIdx.x >> 5;
  const int lane = threadIdx.x & 31;
  const int bh = blockIdx.x;
  const int b = bh >> 4, h = bh & 15;
  const int m0 = (blockIdx.y * 4 + wave) * 16;
  const int hi = lane >> 4, cl = lane & 15;

  const _Float16* qbase = q + (size_t)(b * kN + m0) * kInner + h * kDh;
  const v16h aq0 = load_frag_a(qbase, kInner);
  const v16h aq1 = load_frag_a(qbase + 32, kInner);

  const _Float16* kbase = k + (size_t)(b * kN) * kInner + h * kDh;
  const _Float16* vbase = vt + (size_t)bh * kDh * kN;

  v8f acc[4] = {};
  float mrun[8], lrun[8];
#pragma unroll
  for (int i = 0; i < 8; ++i) { mrun[i] = -3.0e38f; lrun[i] = 0.f; }

  constexpr int TJ = 32;
  constexpr int NT = kN / TJ;   // 64 key tiles
  if (wave == 0) {
    tdm_load_2d(lds_off(&kbuf[0][0]), kbase, 64, TJ, kInner);  // 32 keys x 64 d
    tdm_load_2d(lds_off(&vbuf[0][0]), vbase, TJ, 64, kN);      // 64 d x 32 j
  }
  for (int t = 0; t < NT; ++t) {
    const int cur = t & 1;
    if (wave == 0) {
      if (t + 1 < NT) {
        tdm_load_2d(lds_off(&kbuf[cur ^ 1][0]),
                    kbase + (size_t)(t + 1) * TJ * kInner, 64, TJ, kInner);
        tdm_load_2d(lds_off(&vbuf[cur ^ 1][0]),
                    vbase + (t + 1) * TJ, TJ, 64, kN);
        __builtin_amdgcn_s_wait_tensorcnt(2);
      } else {
        __builtin_amdgcn_s_wait_tensorcnt(0);
      }
    }
    __syncthreads();

    // ---- S = Q K^T : two 16-key column tiles, K(=d) = 64 in two steps
    v8f dts[2] = {};
#pragma unroll
    for (int nt = 0; nt < 2; ++nt) {
      dts[nt] = wmma_f16(aq0, load_frag_b(&kbuf[cur][nt * 16 * 64], 64), dts[nt]);
      dts[nt] = wmma_f16(aq1, load_frag_b(&kbuf[cur][nt * 16 * 64 + 32], 64), dts[nt]);
    }
#pragma unroll
    for (int nt = 0; nt < 2; ++nt)
#pragma unroll
      for (int i = 0; i < 8; ++i) dts[nt][i] *= 0.125f;   // Dh^-0.5

    // ---- online softmax over this 32-key tile (row stats across half-wave)
    float tm[8];
#pragma unroll
    for (int i = 0; i < 8; ++i) {
      float m = fmaxf(dts[0][i], dts[1][i]);
      m = xor_fmax<1>(m); m = xor_fmax<2>(m);
      m = xor_fmax<4>(m); m = xor_fmax<8>(m);
      tm[i] = m;
    }
    float corr[8], tsum[8];
#pragma unroll
    for (int i = 0; i < 8; ++i) {
      const float mn = fmaxf(mrun[i], tm[i]);
      corr[i] = __expf(mrun[i] - mn);
      mrun[i] = mn;
      tsum[i] = 0.f;
    }
#pragma unroll
    for (int nt = 0; nt < 2; ++nt)
#pragma unroll
      for (int i = 0; i < 8; ++i) {
        const float p = __expf(dts[nt][i] - mrun[i]);
        dts[nt][i] = p;
        tsum[i] += p;
      }
#pragma unroll
    for (int i = 0; i < 8; ++i) {
      float s = tsum[i];
      s = xor_fadd<1>(s); s = xor_fadd<2>(s);
      s = xor_fadd<4>(s); s = xor_fadd<8>(s);
      lrun[i] = lrun[i] * corr[i] + s;
    }
#pragma unroll
    for (int nt = 0; nt < 4; ++nt)
#pragma unroll
      for (int i = 0; i < 8; ++i) acc[nt][i] *= corr[i];

    // ---- P (C-layout f32) -> LDS f16 -> A fragment
#pragma unroll
    for (int nt = 0; nt < 2; ++nt)
#pragma unroll
      for (int i = 0; i < 8; ++i)
        pbuf[wave][(i + 8 * hi) * 32 + nt * 16 + cl] = (_Float16)dts[nt][i];
    __syncthreads();
    const v16h ap = load_frag_a(&pbuf[wave][0], 32);

    // ---- O += P V  (vbuf rows are d, K=j contiguous)
#pragma unroll
    for (int dt = 0; dt < 4; ++dt)
      acc[dt] = wmma_f16(ap, load_frag_b(&vbuf[cur][dt * 16 * 32], 32), acc[dt]);
    __syncthreads();   // done with kbuf/vbuf[cur] before TDM refill
  }

#pragma unroll
  for (int dt = 0; dt < 4; ++dt)
#pragma unroll
    for (int i = 0; i < 8; ++i) {
      const int r = m0 + i + 8 * hi;
      out[(size_t)(b * kN + r) * kInner + h * kDh + dt * 16 + cl] =
          (_Float16)(acc[dt][i] / lrun[i]);
    }
}

// ---------------------------------------------------------------- launcher
extern "C" void kernel_launch(void* const* d_in, const int* in_sizes, int n_in,
                              void* d_out, int out_size, void* d_ws, size_t ws_size,
                              hipStream_t stream) {
  const float* x  = (const float*)d_in[0];
  const float* g  = (const float*)d_in[1];
  const float* be = (const float*)d_in[2];
  const float* keysF[4] = {(const float*)d_in[3], (const float*)d_in[5],
                           (const float*)d_in[7], (const float*)d_in[9]};
  const float* valsF[4] = {(const float*)d_in[4], (const float*)d_in[6],
                           (const float*)d_in[8], (const float*)d_in[10]};

  char* w = (char*)d_ws;
  size_t off = 0;
  auto alloc = [&](size_t bytes) -> void* {
    void* p = w + off;
    off += (bytes + 255) & ~(size_t)255;
    return p;
  };
  const size_t mat16 = (size_t)1024 * 1024 * 2;  // 2 MB
  _Float16* xn16 = (_Float16*)alloc((size_t)kM * kDim * 2);
  _Float16* key16[4];
  _Float16* valT16[4];
  for (int i = 0; i < 4; ++i) key16[i]  = (_Float16*)alloc(mat16);
  for (int i = 0; i < 4; ++i) valT16[i] = (_Float16*)alloc(mat16);
  float*    simf   = (float*)alloc((size_t)kM * kP * 4);
  _Float16* attn16 = (_Float16*)alloc((size_t)kM * kP * 2);
  _Float16* q16    = (_Float16*)alloc((size_t)kM * kInner * 2);
  _Float16* k16    = (_Float16*)alloc((size_t)kM * kInner * 2);
  _Float16* vt16   = (_Float16*)alloc((size_t)kM * kInner * 2);
  _Float16* ao16   = (_Float16*)alloc((size_t)kM * kInner * 2);
  if (off > ws_size) return;  // workspace too small; nothing safe to do

  const dim3 gemmGrid(1024 / 64, kM / 64);  // (16, 128)
  const dim3 gemmBlk(128);

  // 1) LayerNorm -> f16
  ln_to_f16_kernel<<<kM, 256, 0, stream>>>(x, g, be, xn16);

  // 2) param matrices -> f16 (keys as-is; vals transposed to [E, P])
  for (int i = 0; i < 4; ++i) {
    cvt_f32_f16_kernel<<<4096, 256, 0, stream>>>(keysF[i], key16[i], 1024 * 1024);
    transpose_cvt_kernel<<<4096, 256, 0, stream>>>(valsF[i], valT16[i]);
  }

  // 3) q/k/v pattention
  _Float16* proj_dst[3] = {q16, k16, vt16};
  for (int p = 0; p < 3; ++p) {
    gemm_nt_kernel<0><<<gemmGrid, gemmBlk, 0, stream>>>(xn16, key16[p], simf);
    l2norm_gelu_kernel<<<kM, 256, 0, stream>>>(simf, attn16);
    if (p < 2)
      gemm_nt_kernel<1><<<gemmGrid, gemmBlk, 0, stream>>>(attn16, valT16[p], proj_dst[p]);
    else
      gemm_nt_kernel<2><<<gemmGrid, gemmBlk, 0, stream>>>(attn16, valT16[p], proj_dst[p]);
  }

  // 4) flash softmax attention: grid (b*h, query tiles of 64 rows / 4 waves)
  flash_attn_kernel<<<dim3(kB * kHeads, kN / 64), 128, 0, stream>>>(q16, k16, vt16, ao16);

  // 5) output pattention -> f32 d_out
  gemm_nt_kernel<0><<<gemmGrid, gemmBlk, 0, stream>>>(ao16, key16[3], simf);
  l2norm_gelu_kernel<<<kM, 256, 0, stream>>>(simf, attn16);
  gemm_nt_kernel<0><<<gemmGrid, gemmBlk, 0, stream>>>(attn16, valT16[3], (float*)d_out);
}
